// CachedSelfAttention_45174466019599
// MI455X (gfx1250) — matrix operand order repaired and checked by reading, not compiled, not run on hardware
//
#include <hip/hip_runtime.h>
#include <hip/hip_bf16.h>

typedef __attribute__((ext_vector_type(16))) _Float16 v16h;
typedef __attribute__((ext_vector_type(8)))  _Float16 v8h;
typedef __attribute__((ext_vector_type(8)))  float    v8f;
typedef __attribute__((ext_vector_type(8)))  float    v8fv;

#define S_NEW    256
#define S_CACHED 8192
#define HD       128
#define DIMC     3072
#define NH       24
#define KSPLIT   4
#define CHUNKS_PER_PART ((S_CACHED + S_NEW) / 32 / KSPLIT)   // 66

// ---------- helpers ----------
// A-fragment per-lane from f32: elems 0..7 = p[0..7], elems 8..15 = p[16..23]
__device__ inline v16h lda_f32_to_f16(const float* __restrict__ p, float scale) {
  v16h r;
#pragma unroll
  for (int i = 0; i < 8; ++i) {
    r[i]     = (_Float16)(p[i]      * scale);
    r[i + 8] = (_Float16)(p[i + 16] * scale);
  }
  return r;
}
__device__ inline v16h ld16_h(const _Float16* p) {
  v16h r;
#pragma unroll
  for (int i = 0; i < 16; ++i) r[i] = p[i + (i & 8)];
  return r;
}
__device__ inline float wave16_max(float v) {
#pragma unroll
  for (int m = 8; m; m >>= 1) v = fmaxf(v, __shfl_xor(v, m, 32));
  return v;
}
__device__ inline float wave16_sum(float v) {
#pragma unroll
  for (int m = 8; m; m >>= 1) v += __shfl_xor(v, m, 32);
  return v;
}

// ---------- elementwise f32 -> f16 (8 elems/thread, vector load/store) ----------
__global__ void __launch_bounds__(256) cvt_f32_f16(
    const float* __restrict__ src, _Float16* __restrict__ dst, int n) {
  const int i = (blockIdx.x * blockDim.x + threadIdx.x) * 8;
  if (i >= n) return;
  v8fv v = *(const v8fv*)(src + i);
  v8h o;
#pragma unroll
  for (int j = 0; j < 8; ++j) o[j] = (_Float16)v[j];
  *(v8h*)(dst + i) = o;
}

// ---------- GEMM: C[M,N] = A[M,K] * W[N,K]^T + bias; all-f16 operands ----------
__global__ void __launch_bounds__(256) gemm_f16(
    const _Float16* __restrict__ A, const _Float16* __restrict__ W,
    const float* __restrict__ bias, float* __restrict__ C,
    int M, int N, int K) {
  const int gw   = (blockIdx.x * blockDim.x + threadIdx.x) >> 5;
  const int lane = threadIdx.x & 31;
  const int tiles_n = N >> 6;
  const int tiles   = (M >> 4) * tiles_n;
  if (gw >= tiles) return;
  const int m0 = (gw / tiles_n) << 4;
  const int n0 = (gw % tiles_n) << 6;
  const int l  = lane & 15;
  const bool hi = lane >= 16;

  v8f acc[4];
  v8f z = {};
#pragma unroll
  for (int j = 0; j < 4; ++j) acc[j] = z;

  const _Float16* arow = A + (size_t)(m0 + l) * K + (hi ? 16 : 0);
  const _Float16* wr[4];
#pragma unroll
  for (int j = 0; j < 4; ++j)
    wr[j] = W + (size_t)(n0 + j * 16 + l) * K + (hi ? 16 : 0);

  for (int k0 = 0; k0 < K; k0 += 32) {
    __builtin_prefetch(arow + k0 + 512, 0, 1);
    // issue all independent loads first: 2x16B (A) + 4x32B (B)
    v8h  alo = *(const v8h*)(arow + k0);
    v8h  ahi = *(const v8h*)(arow + k0 + 8);
    v16h b0  = *(const v16h*)(wr[0] + k0);
    v16h b1  = *(const v16h*)(wr[1] + k0);
    v16h b2  = *(const v16h*)(wr[2] + k0);
    v16h b3  = *(const v16h*)(wr[3] + k0);
    v16h a   = __builtin_shufflevector(alo, ahi, 0, 1, 2, 3, 4, 5, 6, 7,
                                       8, 9, 10, 11, 12, 13, 14, 15);
    acc[0] = __builtin_amdgcn_wmma_f32_16x16x32_f16(false, a, false, b0, (short)0,
                                                    acc[0], false, false);
    acc[1] = __builtin_amdgcn_wmma_f32_16x16x32_f16(false, a, false, b1, (short)0,
                                                    acc[1], false, false);
    acc[2] = __builtin_amdgcn_wmma_f32_16x16x32_f16(false, a, false, b2, (short)0,
                                                    acc[2], false, false);
    acc[3] = __builtin_amdgcn_wmma_f32_16x16x32_f16(false, a, false, b3, (short)0,
                                                    acc[3], false, false);
  }

  const int row0 = m0 + (hi ? 8 : 0);
#pragma unroll
  for (int j = 0; j < 4; ++j) {
    const int col = n0 + j * 16 + l;
    const float bv = bias[col];
#pragma unroll
    for (int r = 0; r < 8; ++r)
      C[(size_t)(row0 + r) * N + col] = acc[j][r] + bv;
  }
}

// ---------- fused RMSNorm (over DIM) + interleaved RoPE, in place ----------
__global__ void __launch_bounds__(256) rmsnorm_rope_kernel(
    float* __restrict__ t, const float* __restrict__ g,
    const float* __restrict__ freqs) {
  const int token = blockIdx.x;
  const int s     = token % S_NEW;
  float* row = t + (size_t)token * DIMC;
  const int tid = threadIdx.x;

  float local[12];
  float ss = 0.f;
#pragma unroll
  for (int i = 0; i < 12; ++i) {
    local[i] = row[tid * 12 + i];
    ss += local[i] * local[i];
  }
  __shared__ float red[256];
  red[tid] = ss;
  __syncthreads();
  for (int off = 128; off > 0; off >>= 1) {
    if (tid < off) red[tid] += red[tid + off];
    __syncthreads();
  }
  const float inv = rsqrtf(red[0] / (float)DIMC + 1e-6f);

#pragma unroll
  for (int i = 0; i < 12; i += 2) {
    const int d0 = tid * 12 + i;
    const float a  = local[i]     * inv * g[d0];
    const float bb = local[i + 1] * inv * g[d0 + 1];
    const int pair = (d0 & (HD - 1)) >> 1;
    const float ang = freqs[s * (HD / 2) + pair];
    const float cs = __cosf(ang), sn = __sinf(ang);
    row[d0]     = a * cs - bb * sn;
    row[d0 + 1] = a * sn + bb * cs;
  }
}

// ---------- attention tile source select (chunk-aligned cache/new split) ----------
__device__ inline void tile_src(int g, int b,
                                const float* kc, const float* vc,
                                const float* kn, const float* vn,
                                const float*& ks, const float*& vs, int& rowbase) {
  if (g < S_CACHED / 32) { ks = kc; vs = vc; rowbase = b * S_CACHED + g * 32; }
  else                   { ks = kn; vs = vn; rowbase = b * S_NEW + (g - S_CACHED / 32) * 32; }
}

// Async-copy one 32x128 f32 tile of K and V into LDS (4 async ops per wave).
__device__ inline void issue_tile(const float* ks, const float* vs, int rowbase, int h,
                                  float* kdst, float* vdst, int tid) {
#pragma unroll
  for (int j = 0; j < 2; ++j) {
    const int i   = tid + j * 512;      // 1024 x 16B chunks per 16KB tile
    const int key = i >> 5;
    const int off = (i & 31) * 16;
    const char* kg = (const char*)(ks + (size_t)(rowbase + key) * DIMC + h * HD) + off;
    const unsigned kl = (unsigned)(size_t)kdst + (unsigned)(i * 16);
    asm volatile("global_load_async_to_lds_b128 %0, %1, off"
                 :: "v"(kl), "v"((unsigned long long)(size_t)kg) : "memory");
    const char* vg = (const char*)(vs + (size_t)(rowbase + key) * DIMC + h * HD) + off;
    const unsigned vl = (unsigned)(size_t)vdst + (unsigned)(i * 16);
    asm volatile("global_load_async_to_lds_b128 %0, %1, off"
                 :: "v"(vl), "v"((unsigned long long)(size_t)vg) : "memory");
  }
}

// ---------- cooperative flash attention: 512 threads = 16 waves, one (b,h,part) per block
__global__ void __launch_bounds__(512) attn_kernel(
    const float* __restrict__ q,      const float* __restrict__ knew,
    const float* __restrict__ vnew,   const float* __restrict__ kcache,
    const float* __restrict__ vcache, float* __restrict__ opart,
    float* __restrict__ mlpart) {
  const int tid  = threadIdx.x;
  const int lane = tid & 31;
  const int wib  = tid >> 5;            // 0..15, one 16-query tile per wave
  const int b    = blockIdx.x / (NH * KSPLIT);
  const int rem  = blockIdx.x % (NH * KSPLIT);
  const int h    = rem / KSPLIT;
  const int part = rem % KSPLIT;
  const int m0   = wib * 16;
  const int l    = lane & 15;
  const bool hi  = lane >= 16;

  __shared__ float    kraw[2][32][128];   // raw f32 tiles (async landing zone)
  __shared__ float    vraw[2][32][128];
  __shared__ _Float16 kt16[2][32][128];   // K tile f16, [key][dim]
  __shared__ _Float16 vt16[2][128][32];   // V tile f16 transposed, [dim][key]
  __shared__ _Float16 pbuf[16][512];      // per-wave P staging (D-layout -> A-layout)
  _Float16* myp = pbuf[wib];

  const float scale = 0.088388347648318f; // 1/sqrt(128)

  // Q fragments: 16x128 as 4 chunks of 16x32
  v16h qf[4];
  {
    const float* qrow = q + (size_t)(b * S_NEW + m0 + l) * DIMC + h * HD + (hi ? 8 : 0);
#pragma unroll
    for (int c = 0; c < 4; ++c) qf[c] = lda_f32_to_f16(qrow + c * 32, scale);
  }

  v8f o[8];
  v8f z = {};
#pragma unroll
  for (int i = 0; i < 8; ++i) o[i] = z;
  float mrow[8], lrow[8];
#pragma unroll
  for (int r = 0; r < 8; ++r) { mrow[r] = -1e30f; lrow[r] = 0.f; }

  // prologue: prefetch tiles 0 and 1 via async DMA
  {
    const float* ks; const float* vs; int rb;
    tile_src(part * CHUNKS_PER_PART + 0, b, kcache, vcache, knew, vnew, ks, vs, rb);
    issue_tile(ks, vs, rb, h, &kraw[0][0][0], &vraw[0][0][0], tid);
    tile_src(part * CHUNKS_PER_PART + 1, b, kcache, vcache, knew, vnew, ks, vs, rb);
    issue_tile(ks, vs, rb, h, &kraw[1][0][0], &vraw[1][0][0], tid);
  }
  asm volatile("s_wait_asynccnt 0x4" ::: "memory");   // tile 0 landed (own ops)
  __syncthreads();                                     // everyone's tile 0 landed

  for (int c = 0; c < CHUNKS_PER_PART; ++c) {
    const int buf = c & 1;

    // convert raw f32 -> f16 tiles (V transposed): tid -> key = tid>>4, d0 = (tid&15)*8
    {
      const int key = tid >> 4;
      const int d0  = (tid & 15) * 8;
#pragma unroll
      for (int i = 0; i < 8; ++i) {
        kt16[buf][key][d0 + i]  = (_Float16)kraw[buf][key][d0 + i];
        vt16[buf][d0 + i][key]  = (_Float16)vraw[buf][key][d0 + i];
      }
    }
    __syncthreads();   // f16 tiles visible

    // ---- scores: S[16q x 32k] = two 16x16 tiles, K=128 over 4 WMMA steps
    v8f s0 = z, s1 = z;
#pragma unroll
    for (int c4 = 0; c4 < 4; ++c4) {
      v16h kb0 = ld16_h(&kt16[buf][l][c4 * 32 + (hi ? 8 : 0)]);
      s0 = __builtin_amdgcn_wmma_f32_16x16x32_f16(false, qf[c4], false, kb0, (short)0,
                                                  s0, false, false);
      v16h kb1 = ld16_h(&kt16[buf][16 + l][c4 * 32 + (hi ? 8 : 0)]);
      s1 = __builtin_amdgcn_wmma_f32_16x16x32_f16(false, qf[c4], false, kb1, (short)0,
                                                  s1, false, false);
    }

    // ---- online softmax per row
    float alpha[8];
#pragma unroll
    for (int r = 0; r < 8; ++r) {
      float smax = wave16_max(fmaxf(s0[r], s1[r]));
      float mnew = fmaxf(mrow[r], smax);
      float a  = __expf(mrow[r] - mnew);
      float p0 = __expf(s0[r] - mnew);
      float p1 = __expf(s1[r] - mnew);
      float ls = wave16_sum(p0 + p1);
      lrow[r] = lrow[r] * a + ls;
      mrow[r] = mnew;
      alpha[r] = a;
      const int rowi = (hi ? 8 : 0) + r;
      myp[rowi * 32 + l]      = (_Float16)p0;
      myp[rowi * 32 + 16 + l] = (_Float16)p1;
    }
#pragma unroll
    for (int nt = 0; nt < 8; ++nt)
#pragma unroll
      for (int r = 0; r < 8; ++r) o[nt][r] *= alpha[r];

    asm volatile("" ::: "memory");
    v16h pa;   // reload P in A-fragment layout (DS ops in-order per wave)
    {
      const _Float16* pr = myp + l * 32 + (hi ? 8 : 0);
#pragma unroll
      for (int i = 0; i < 8; ++i) { pa[i] = pr[i]; pa[i + 8] = pr[i + 16]; }
    }
    asm volatile("" ::: "memory");

    // ---- O += P[16x32] * V[32x128]; V B-frags contiguous from transposed tile
#pragma unroll
    for (int nt = 0; nt < 8; ++nt) {
      v16h vb = ld16_h(&vt16[buf][nt * 16 + l][hi ? 8 : 0]);
      o[nt] = __builtin_amdgcn_wmma_f32_16x16x32_f16(false, pa, false, vb, (short)0,
                                                     o[nt], false, false);
    }

    // ---- prefetch tile c+2 into the raw buffer we just consumed
    if (c + 2 < CHUNKS_PER_PART) {
      const float* ks; const float* vs; int rb;
      tile_src(part * CHUNKS_PER_PART + c + 2, b, kcache, vcache, knew, vnew, ks, vs, rb);
      issue_tile(ks, vs, rb, h, &kraw[buf][0][0], &vraw[buf][0][0], tid);
      asm volatile("s_wait_asynccnt 0x4" ::: "memory");  // tile c+1 landed
    } else {
      asm volatile("s_wait_asynccnt 0x0" ::: "memory");
    }
    __syncthreads();   // everyone's tile c+1 landed; raw[buf] safe to rewrite next iter
  }

  // ---- write partials (unnormalized O relative to mrow, plus per-row m,l)
  const size_t base = (size_t)((part * 2 + b) * NH + h) * S_NEW;
#pragma unroll
  for (int nt = 0; nt < 8; ++nt) {
    const int col = nt * 16 + l;
#pragma unroll
    for (int r = 0; r < 8; ++r) {
      const int row = m0 + (hi ? 8 : 0) + r;
      opart[(base + row) * HD + col] = o[nt][r];
    }
  }
  if (l == 0) {
#pragma unroll
    for (int r = 0; r < 8; ++r) {
      const size_t idx = (base + m0 + (hi ? 8 : 0) + r) * 2;
      mlpart[idx]     = mrow[r];
      mlpart[idx + 1] = lrow[r];
    }
  }
}

// ---------- combine KSPLIT partials: one wave per (b,h,q) ----------
__global__ void __launch_bounds__(256) combine_kernel(
    const float* __restrict__ opart, const float* __restrict__ ml,
    float* __restrict__ out) {
  const int gw   = (blockIdx.x * blockDim.x + threadIdx.x) >> 5;
  const int lane = threadIdx.x & 31;
  if (gw >= 2 * NH * S_NEW) return;
  const int b  = gw / (NH * S_NEW);
  const int h  = (gw / S_NEW) % NH;
  const int qi = gw % S_NEW;

  float mi[KSPLIT], li[KSPLIT], w[KSPLIT];
  float m = -1e30f;
#pragma unroll
  for (int p = 0; p < KSPLIT; ++p) {
    const size_t idx = ((size_t)((p * 2 + b) * NH + h) * S_NEW + qi) * 2;
    mi[p] = ml[idx]; li[p] = ml[idx + 1];
    m = fmaxf(m, mi[p]);
  }
  float denom = 0.f;
#pragma unroll
  for (int p = 0; p < KSPLIT; ++p) { w[p] = __expf(mi[p] - m); denom += w[p] * li[p]; }
  const float inv = 1.0f / denom;

#pragma unroll
  for (int j = 0; j < 4; ++j) {
    const int d = lane + j * 32;
    float acc = 0.f;
#pragma unroll
    for (int p = 0; p < KSPLIT; ++p)
      acc += opart[((size_t)((p * 2 + b) * NH + h) * S_NEW + qi) * HD + d] * w[p];
    out[(size_t)(b * S_NEW + qi) * DIMC + h * HD + d] = acc * inv;
  }
}

// ---------- host ----------
extern "C" void kernel_launch(void* const* d_in, const int* in_sizes, int n_in,
                              void* d_out, int out_size, void* d_ws, size_t ws_size,
                              hipStream_t stream) {
  (void)in_sizes; (void)n_in; (void)out_size; (void)ws_size;
  const float* x      = (const float*)d_in[0];
  const float* freqs  = (const float*)d_in[1];
  const float* kcache = (const float*)d_in[2];
  const float* vcache = (const float*)d_in[3];
  const float* Wq = (const float*)d_in[4];
  const float* bq = (const float*)d_in[5];
  const float* Wk = (const float*)d_in[6];
  const float* bk = (const float*)d_in[7];
  const float* Wv = (const float*)d_in[8];
  const float* bv = (const float*)d_in[9];
  const float* Wo = (const float*)d_in[10];
  const float* bo = (const float*)d_in[11];
  const float* gq = (const float*)d_in[12];
  const float* gk = (const float*)d_in[13];
  float* out = (float*)d_out;

  const size_t T  = (size_t)(2 * S_NEW) * DIMC;      // 512*3072 activation elems
  const size_t WE = (size_t)DIMC * DIMC;             // 3072*3072 weight elems

  float* qb     = (float*)d_ws;
  float* kb     = qb + T;
  float* vb     = kb + T;
  float* ab     = vb + T;
  float* opart  = ab + T;                            // KSPLIT * T floats
  float* mlpart = opart + (size_t)KSPLIT * T;        // KSPLIT*2*NH*S_NEW*2 floats
  _Float16* xh  = (_Float16*)(mlpart + (size_t)KSPLIT * 2 * NH * S_NEW * 2);
  _Float16* abh = xh + T;
  _Float16* Wqh = abh + T;
  _Float16* Wkh = Wqh + WE;
  _Float16* Wvh = Wkh + WE;
  _Float16* Woh = Wvh + WE;

  const int M = 2 * S_NEW;
  const dim3 blk(256);
  const dim3 cvtA_grid((unsigned)(T / 8 / 256));     // 768
  const dim3 cvtW_grid((unsigned)(WE / 8 / 256));    // 4608
  const dim3 gemm_grid((M / 16) * (DIMC / 64) / 8);  // 192 blocks, 8 waves each

  // one-time precision conversion (weights resident in L2 afterwards)
  cvt_f32_f16<<<cvtA_grid, blk, 0, stream>>>(x,  xh,  (int)T);
  cvt_f32_f16<<<cvtW_grid, blk, 0, stream>>>(Wq, Wqh, (int)WE);
  cvt_f32_f16<<<cvtW_grid, blk, 0, stream>>>(Wk, Wkh, (int)WE);
  cvt_f32_f16<<<cvtW_grid, blk, 0, stream>>>(Wv, Wvh, (int)WE);
  cvt_f32_f16<<<cvtW_grid, blk, 0, stream>>>(Wo, Woh, (int)WE);

  gemm_f16<<<gemm_grid, blk, 0, stream>>>(xh, Wqh, bq, qb, M, DIMC, DIMC);
  gemm_f16<<<gemm_grid, blk, 0, stream>>>(xh, Wkh, bk, kb, M, DIMC, DIMC);
  gemm_f16<<<gemm_grid, blk, 0, stream>>>(xh, Wvh, bv, vb, M, DIMC, DIMC);

  rmsnorm_rope_kernel<<<dim3(M), blk, 0, stream>>>(qb, gq, freqs);
  rmsnorm_rope_kernel<<<dim3(M), blk, 0, stream>>>(kb, gk, freqs);

  attn_kernel<<<dim3(2 * NH * KSPLIT), dim3(512), 0, stream>>>(
      qb, kb, vb, kcache, vcache, opart, mlpart);

  combine_kernel<<<dim3(2 * NH * S_NEW / 8), blk, 0, stream>>>(opart, mlpart, ab);

  cvt_f32_f16<<<cvtA_grid, blk, 0, stream>>>(ab, abh, (int)T);
  gemm_f16<<<gemm_grid, blk, 0, stream>>>(abh, Woh, bo, out, M, DIMC, DIMC);
}
